// EnTransformer_26706106646673
// MI455X (gfx1250) — matrix operand order repaired and checked by reading, not compiled, not run on hardware
//
#include <hip/hip_runtime.h>

// ---------------------------------------------------------------------------
// EnTransformer for MI455X (gfx1250, wave32, WMMA bf16 16x16x32, f32 accum)
// B=8 N=256 V=8 F=128 H=4 NB=8 L=3 MLPD=128, E=N*(N-1)=65280 per batch
// ---------------------------------------------------------------------------

#define DEVI __device__ __forceinline__

typedef __bf16 v16bf __attribute__((ext_vector_type(16)));
typedef float  v8f   __attribute__((ext_vector_type(8)));

union FragAB {
  v16bf v;
  float4 f4[2];
};

DEVI unsigned short f32_to_bf16(float x) {
  unsigned int u = __builtin_bit_cast(unsigned int, x);
  unsigned int r = u + 0x7FFFu + ((u >> 16) & 1u);   // round-to-nearest-even
  return (unsigned short)(r >> 16);
}

DEVI float seluf(float x) {
  const float sc = 1.0507009873554805f, al = 1.6732632423543772f;
  return x > 0.f ? sc * x : sc * al * (__expf(x) - 1.f);
}

// A-fragment (16x32 bf16): lane: M=lane&15, half=lane>>4.
// element t -> K = (t&7) + (t>>3)*16 + half*8  => two contiguous 16B runs.
DEVI v16bf load_a_frag(const unsigned short* slice, int S, int k0, int lane) {
  int row = lane & 15, half = lane >> 4;
  const unsigned short* p = slice + row * S + k0 + half * 8;
  FragAB f;
  f.f4[0] = *(const float4*)(p);
  f.f4[1] = *(const float4*)(p + 16);
  return f.v;
}

// B-fragment (32x16 bf16): lane: N=lane&15, half=lane>>4.
// element t -> K = k0 + (t&7) + (t>>3)*16 + half*8  => two contiguous 16B runs.
// Wt stored column-major: Wt[col*Ktot + k], bf16 bits in ushort.
DEVI v16bf load_b_frag(const unsigned short* __restrict__ Wt, int Ktot,
                       int ntile, int k0, int lane) {
  int col = ntile * 16 + (lane & 15), half = lane >> 4;
  const unsigned short* p = Wt + col * Ktot + k0 + half * 8;
  FragAB f;
  f.f4[0] = *(const float4*)(p);
  f.f4[1] = *(const float4*)(p + 16);
  return f.v;
}

DEVI v8f wmma_bf16(v16bf a, v16bf b, v8f c) {
  return __builtin_amdgcn_wmma_f32_16x16x32_bf16(false, a, false, b,
                                                 (short)0, c, false, false);
}

// ---------------------------------------------------------------------------
// Weight prep: convert + transpose (column-major, K contiguous) to bf16 in ws.
// ---------------------------------------------------------------------------
__global__ void k_prep(const float* __restrict__ W1, const float* __restrict__ W2,
                       const float* __restrict__ Wl, const float* __restrict__ Wmix,
                       const float* __restrict__ Wmsg, const float* __restrict__ Wfs,
                       unsigned short* __restrict__ W1ac, unsigned short* __restrict__ W2t,
                       unsigned short* __restrict__ W3t, unsigned short* __restrict__ Wmsgt,
                       unsigned short* __restrict__ Wfst) {
  int idx = blockIdx.x * blockDim.x + threadIdx.x;
  const int n0 = 3 * 128 * 192, n1 = 3 * 128 * 128, n2 = 3 * 80 * 128,
            n3 = 3 * 128 * 128, n4 = 128 * 128;
  if (idx < n0) { // W1ac[l][col][k]: k<128 -> W1 send rows; k>=128 -> rad rows 256..319
    int l = idx / (128 * 192), rem = idx % (128 * 192);
    int col = rem / 192, k = rem % 192;
    float s = (k < 128) ? W1[(l * 320 + k) * 128 + col]
                        : W1[(l * 320 + 256 + (k - 128)) * 128 + col];
    W1ac[idx] = f32_to_bf16(s);
    return;
  }
  idx -= n0;
  if (idx < n1) {
    int l = idx / 16384, rem = idx % 16384, col = rem / 128, k = rem % 128;
    W2t[idx] = f32_to_bf16(W2[(l * 128 + k) * 128 + col]);
    return;
  }
  idx -= n1;
  if (idx < n2) { // W3t: cols 0..3 = Wl, 4..15 = 0, 16..79 = Wmix
    int l = idx / 10240, rem = idx % 10240, col = rem / 128, k = rem % 128;
    float s = 0.f;
    if (col < 16) { if (col < 4) s = Wl[(l * 128 + k) * 4 + col]; }
    else          s = Wmix[(l * 128 + k) * 64 + (col - 16)];
    W3t[idx] = f32_to_bf16(s);
    return;
  }
  idx -= n2;
  if (idx < n3) {
    int l = idx / 16384, rem = idx % 16384, col = rem / 128, k = rem % 128;
    Wmsgt[idx] = f32_to_bf16(Wmsg[(l * 128 + k) * 128 + col]);
    return;
  }
  idx -= n3;
  if (idx < n4) {
    int col = idx / 128, k = idx % 128;
    Wfst[idx] = f32_to_bf16(Wfs[k * 128 + col]);
  }
}

// h0 = 1, pos0 = broadcast x over V channels
__global__ void k_init(const float* __restrict__ x, float* __restrict__ h0,
                       float* __restrict__ pos0) {
  int idx = blockIdx.x * blockDim.x + threadIdx.x;
  if (idx < 8 * 256 * 128) h0[idx] = 1.f;
  if (idx < 8 * 256 * 8 * 3) {
    int node = idx / 24, c = idx % 3;
    pos0[idx] = x[node * 3 + c];
  }
}

// ---------------------------------------------------------------------------
// Generic [R,128] x [128,128] bf16-WMMA GEMM (R multiple of 64). 4 waves/WG,
// one 16-row tile per wave. Used for hs = h@Wmsg and scal_out = h@Wfin_s.
// ---------------------------------------------------------------------------
__global__ __launch_bounds__(128) void k_rows_gemm(
    const float* __restrict__ A, const unsigned short* __restrict__ Wt,
    float* __restrict__ out, int R) {
  __shared__ alignas(16) unsigned short sl[4][16 * 128];
  int t = threadIdx.x, wave = t >> 5, lane = t & 31;
  int rowbase = blockIdx.x * 64 + wave * 16;
  if (rowbase >= R) return;
  int row = lane & 15, half = lane >> 4;
  int rg = rowbase + row;
  // stage 16 rows f32 -> bf16 LDS
  for (int q = 0; q < 16; q++) {
    int c = half * 64 + q * 4;
    float4 v = *(const float4*)(A + rg * 128 + c);
    unsigned short* d = &sl[wave][row * 128 + c];
    d[0] = f32_to_bf16(v.x); d[1] = f32_to_bf16(v.y);
    d[2] = f32_to_bf16(v.z); d[3] = f32_to_bf16(v.w);
  }
  v8f acc[8];
#pragma unroll
  for (int n = 0; n < 8; n++)
#pragma unroll
    for (int r = 0; r < 8; r++) acc[n][r] = 0.f;
#pragma unroll
  for (int k0 = 0; k0 < 128; k0 += 32) {
    v16bf a = load_a_frag(sl[wave], 128, k0, lane);
#pragma unroll
    for (int n = 0; n < 8; n++) {
      v16bf b = load_b_frag(Wt, 128, n, k0, lane);
      acc[n] = wmma_bf16(a, b, acc[n]);
    }
  }
#pragma unroll
  for (int n = 0; n < 8; n++) {
    int col = n * 16 + row;
#pragma unroll
    for (int r = 0; r < 8; r++)
      out[(rowbase + r + half * 8) * 128 + col] = acc[n][r];
  }
}

// ---------------------------------------------------------------------------
// Fused edge kernel: one WG per (batch, receiver i). 4 waves x 4 row-tiles
// cover the 255 edges. feat -> GEMM1(K=192) -> selu -> GEMM2 -> selu ->
// GEMM3 (logits+mix) -> softmax over j -> agg_s/agg_v -> node MLP -> update.
// ---------------------------------------------------------------------------
__global__ __launch_bounds__(128) void k_edge(
    const float* __restrict__ pos_in, const float* __restrict__ h_in,
    const float* __restrict__ hs,
    const unsigned short* __restrict__ W1ac,  // [128 cols][192]
    const unsigned short* __restrict__ W2t,   // [128 cols][128]
    const unsigned short* __restrict__ W3t,   // [ 80 cols][128]
    const float* __restrict__ W1f,            // this layer's full W1 [320][128]
    const float* __restrict__ b1, const float* __restrict__ b2,
    const float* __restrict__ Wo1, const float* __restrict__ bo1,
    const float* __restrict__ Wo2, const float* __restrict__ bo2,
    const float* __restrict__ Wo3, const float* __restrict__ bo3,
    float* __restrict__ pos_out, float* __restrict__ h_out) {
  __shared__ alignas(16) unsigned short feat[4][16 * 200]; // per-wave feat/m slice (bf16 bits)
  __shared__ float shs[4][16 * 24];   // per-wave sh1 (sqrt3*u) [16 rows][8 vec][3]
  __shared__ float msgv[256 * 24];    // msg_v per edge [v][c]
  __shared__ float aL[256 * 4];       // logits -> attention weights
  __shared__ float c1s[128], cb2[128], aggs[128], invb[2][128];

  int bi = blockIdx.x;
  int b = bi >> 8, i = bi & 255;
  int t = threadIdx.x, wave = t >> 5, lane = t & 31;
  int ln15 = lane & 15, half = lane >> 4;
  const float* hrow = h_in + (b * 256 + i) * 128;

  // c1 = h_recv @ W1b + b1 (shared across all 255 edges); stage b2
  {
    float s = b1[t];
    for (int k = 0; k < 128; k++) s += hrow[k] * W1f[(128 + k) * 128 + t];
    c1s[t] = s;
    cb2[t] = b2[t];
  }
  __syncthreads();

  for (int tt = 0; tt < 4; tt++) {
    int rt = wave * 4 + tt;
    int j = rt * 16 + ln15;            // edge index handled by this lane pair
    unsigned short* fsl = feat[wave];
    float* ssl = shs[wave];

    // ---- build feat row: [h_send(0..127) | rad(128..191)], sh1 to ssl ----
    if (j < 255) {
      int send = i + 1 + j; if (send >= 256) send -= 256;
      const float* hsrow = h_in + (b * 256 + send) * 128;
      for (int q = 0; q < 16; q++) {
        int c = half * 64 + q * 4;
        float4 v4 = *(const float4*)(hsrow + c);
        unsigned short* d = fsl + ln15 * 200 + c;
        d[0] = f32_to_bf16(v4.x); d[1] = f32_to_bf16(v4.y);
        d[2] = f32_to_bf16(v4.z); d[3] = f32_to_bf16(v4.w);
      }
      for (int vv = 0; vv < 4; vv++) {
        int v = half * 4 + vv;
        const float* ps = pos_in + ((b * 256 + send) * 8 + v) * 3;
        const float* pr = pos_in + ((b * 256 + i) * 8 + v) * 3;
        float dx = ps[0] - pr[0], dy = ps[1] - pr[1], dz = ps[2] - pr[2];
        float l2 = dx * dx + dy * dy + dz * dz;
        float r = l2 > 0.f ? __fsqrt_rn(l2) : 0.f;
        float ir = r > 0.f ? 1.f / r : 0.f;
        const float s3 = 1.7320508075688772f;
        ssl[ln15 * 24 + v * 3 + 0] = s3 * dx * ir;
        ssl[ln15 * 24 + v * 3 + 1] = s3 * dy * ir;
        ssl[ln15 * 24 + v * 3 + 2] = s3 * dz * ir;
        // rad = bessel(r) * soft_envelope(r)
        float xarg = 2.f * (1.f - r * 0.1f);
        float sus = xarg > 0.f ? __expf(-1.f / xarg) : 0.f;
        float env = 1.2f * 1.6487212707001282f * sus;       // 1.2*e^0.5 * sus
        const float cn = 0.4472135954999579f;               // sqrt(2/RMAX)
        for (int k = 1; k <= 8; k++) {
          float kk = (float)k * 3.14159265358979323846f;
          float val = (r > 0.f) ? (__sinf(kk * r * 0.1f) / r) : (kk * 0.1f);
          fsl[ln15 * 200 + 128 + v * 8 + (k - 1)] = f32_to_bf16(cn * val * env);
        }
      }
    } else {                                   // pad row 255
      for (int c = 0; c < 96; c++) fsl[ln15 * 200 + half * 96 + c] = 0;
      for (int c = 0; c < 12; c++) ssl[ln15 * 24 + half * 12 + c] = 0.f;
    }

    // ---- GEMM1: m1 = feat[16,192] @ W1ac + c1 ----
    v8f acc[8];
#pragma unroll
    for (int n = 0; n < 8; n++) {
      float cv = c1s[n * 16 + ln15];
#pragma unroll
      for (int r = 0; r < 8; r++) acc[n][r] = cv;
    }
#pragma unroll
    for (int k0 = 0; k0 < 192; k0 += 32) {
      v16bf a = load_a_frag(fsl, 200, k0, lane);
      __builtin_prefetch(W1ac + ln15 * 192 + k0, 0, 1);
#pragma unroll
      for (int n = 0; n < 8; n++)
        acc[n] = wmma_bf16(a, load_b_frag(W1ac, 192, n, k0, lane), acc[n]);
    }
#pragma unroll
    for (int n = 0; n < 8; n++) {
      int col = n * 16 + ln15;
#pragma unroll
      for (int r = 0; r < 8; r++)
        fsl[(r + half * 8) * 200 + col] = f32_to_bf16(seluf(acc[n][r]));
    }

    // ---- GEMM2: m2 = m1 @ W2 + b2 ----
#pragma unroll
    for (int n = 0; n < 8; n++) {
      float cv = cb2[n * 16 + ln15];
#pragma unroll
      for (int r = 0; r < 8; r++) acc[n][r] = cv;
    }
#pragma unroll
    for (int k0 = 0; k0 < 128; k0 += 32) {
      v16bf a = load_a_frag(fsl, 200, k0, lane);
#pragma unroll
      for (int n = 0; n < 8; n++)
        acc[n] = wmma_bf16(a, load_b_frag(W2t, 128, n, k0, lane), acc[n]);
    }
#pragma unroll
    for (int n = 0; n < 8; n++) {
      int col = n * 16 + ln15;
#pragma unroll
      for (int r = 0; r < 8; r++)
        fsl[(r + half * 8) * 200 + col] = f32_to_bf16(seluf(acc[n][r]));
    }

    // ---- GEMM3: [logits | mix] = m2 @ W3 (cols 0..3 = Wl, 16..79 = Wmix) ----
    v8f a3[5];
#pragma unroll
    for (int n = 0; n < 5; n++)
#pragma unroll
      for (int r = 0; r < 8; r++) a3[n][r] = 0.f;
#pragma unroll
    for (int k0 = 0; k0 < 128; k0 += 32) {
      v16bf a = load_a_frag(fsl, 200, k0, lane);
#pragma unroll
      for (int n = 0; n < 5; n++)
        a3[n] = wmma_bf16(a, load_b_frag(W3t, 128, n, k0, lane), a3[n]);
    }
    if (ln15 < 4) {
#pragma unroll
      for (int r = 0; r < 8; r++)
        aL[(rt * 16 + r + half * 8) * 4 + ln15] = a3[0][r];
    }
    // mix -> msg_v: butterfly-reduce over the 8 w-lanes sharing (half, v)
#pragma unroll
    for (int n = 1; n < 5; n++) {
      int colm = (n - 1) * 16 + ln15;   // 0..63
      int v = colm >> 3, w = colm & 7;  // w == lane&7
#pragma unroll
      for (int r = 0; r < 8; r++) {
        int lr = r + half * 8;
        float mval = a3[n][r];
#pragma unroll
        for (int c = 0; c < 3; c++) {
          float p = mval * ssl[lr * 24 + w * 3 + c];
          p += __shfl_xor(p, 1, 32);
          p += __shfl_xor(p, 2, 32);
          p += __shfl_xor(p, 4, 32);
          if (w == 0) msgv[(rt * 16 + lr) * 24 + v * 3 + c] = p;
        }
      }
    }
  }
  __syncthreads();

  // ---- softmax over j (per head; wave w handles head w) ----
  {
    int hh = wave;
    float mx = -1e30f;
    for (int j = lane; j < 255; j += 32) mx = fmaxf(mx, aL[j * 4 + hh]);
    for (int o = 16; o; o >>= 1) mx = fmaxf(mx, __shfl_xor(mx, o, 32));
    float sm = 0.f;
    for (int j = lane; j < 255; j += 32) sm += __expf(aL[j * 4 + hh] - mx);
    for (int o = 16; o; o >>= 1) sm += __shfl_xor(sm, o, 32);
    float is = 1.f / sm;
    for (int j = lane; j < 255; j += 32)
      aL[j * 4 + hh] = __expf(aL[j * 4 + hh] - mx) * is;
    if (lane == 0) aL[255 * 4 + hh] = 0.f;
  }
  __syncthreads();

  // ---- aggregation ----
  if (t < 24) {           // agg_v -> pos update
    int v = t / 3, hh = v >> 1;
    float s = 0.f;
    for (int j = 0; j < 255; j++) s += aL[j * 4 + hh] * msgv[j * 24 + t];
    int o = ((b * 256 + i) * 8 + v) * 3 + (t % 3);
    pos_out[o] = pos_in[o] + s;
  }
  {                       // agg_s
    int c = t, hh = c >> 5;
    float s = 0.f;
    const float* hsb = hs + b * 256 * 128;
    int send = i + 1;
    for (int j = 0; j < 255; j++) {
      if (send >= 256) send -= 256;
      s += aL[j * 4 + hh] * hsb[send * 128 + c];
      send++;
    }
    aggs[c] = s;
  }
  __syncthreads();

  // ---- node MLP: h += Wo3(selu(Wo2(selu(Wo1(agg_s))))) ----
  {
    float s = bo1[t];
    for (int k = 0; k < 128; k++) s += aggs[k] * Wo1[k * 128 + t];
    invb[0][t] = seluf(s);
  }
  __syncthreads();
  {
    float s = bo2[t];
    for (int k = 0; k < 128; k++) s += invb[0][k] * Wo2[k * 128 + t];
    invb[1][t] = seluf(s);
  }
  __syncthreads();
  {
    float s = bo3[t];
    for (int k = 0; k < 128; k++) s += invb[1][k] * Wo3[k * 128 + t];
    h_out[(b * 256 + i) * 128 + t] = hrow[t] + s;
  }
}

// vec_out = (pos - com) @ Wfin_v + com   (one thread per node)
__global__ void k_vec(const float* __restrict__ pos, const float* __restrict__ Wv,
                      float* __restrict__ out) {
  int n = blockIdx.x * blockDim.x + threadIdx.x;
  if (n >= 8 * 256) return;
  const float* p = pos + n * 24;
  float com[3] = {0.f, 0.f, 0.f};
  for (int v = 0; v < 8; v++)
    for (int c = 0; c < 3; c++) com[c] += p[v * 3 + c];
  com[0] *= 0.125f; com[1] *= 0.125f; com[2] *= 0.125f;
  for (int w = 0; w < 8; w++)
    for (int c = 0; c < 3; c++) {
      float s = com[c];
      for (int v = 0; v < 8; v++) s += (p[v * 3 + c] - com[c]) * Wv[v * 8 + w];
      out[n * 24 + w * 3 + c] = s;
    }
}

// ---------------------------------------------------------------------------
extern "C" void kernel_launch(void* const* d_in, const int* in_sizes, int n_in,
                              void* d_out, int out_size, void* d_ws, size_t ws_size,
                              hipStream_t stream) {
  (void)in_sizes; (void)n_in; (void)out_size; (void)ws_size;
  const float* x    = (const float*)d_in[0];
  const float* W1   = (const float*)d_in[1];
  const float* b1   = (const float*)d_in[2];
  const float* W2   = (const float*)d_in[3];
  const float* b2   = (const float*)d_in[4];
  const float* Wl   = (const float*)d_in[5];
  const float* Wmix = (const float*)d_in[6];
  const float* Wmsg = (const float*)d_in[7];
  const float* Wo1  = (const float*)d_in[8];
  const float* bo1  = (const float*)d_in[9];
  const float* Wo2  = (const float*)d_in[10];
  const float* bo2  = (const float*)d_in[11];
  const float* Wo3  = (const float*)d_in[12];
  const float* bo3  = (const float*)d_in[13];
  const float* Wfs  = (const float*)d_in[14];
  const float* Wfv  = (const float*)d_in[15];

  // workspace carve (~3.98 MB, all 256B-aligned chunk sizes)
  char* ws = (char*)d_ws;
  unsigned short* W1ac  = (unsigned short*)ws; ws += 3 * 128 * 192 * 2;
  unsigned short* W2t   = (unsigned short*)ws; ws += 3 * 128 * 128 * 2;
  unsigned short* W3t   = (unsigned short*)ws; ws += 3 * 80 * 128 * 2;
  unsigned short* Wmsgt = (unsigned short*)ws; ws += 3 * 128 * 128 * 2;
  unsigned short* Wfst  = (unsigned short*)ws; ws += 128 * 128 * 2;
  float* hA = (float*)ws; ws += 8 * 256 * 128 * 4;
  float* hB = (float*)ws; ws += 8 * 256 * 128 * 4;
  float* pA = (float*)ws; ws += 8 * 256 * 24 * 4;
  float* pB = (float*)ws; ws += 8 * 256 * 24 * 4;
  float* hsb = (float*)ws; ws += 8 * 256 * 128 * 4;

  k_prep<<<(219136 + 255) / 256, 256, 0, stream>>>(W1, W2, Wl, Wmix, Wmsg, Wfs,
                                                   W1ac, W2t, W3t, Wmsgt, Wfst);
  k_init<<<(8 * 256 * 128 + 255) / 256, 256, 0, stream>>>(x, hA, pA);

  float* hbuf[2] = {hA, hB};
  float* pbuf[2] = {pA, pB};
  for (int l = 0; l < 3; l++) {
    const float* h_in = hbuf[l & 1];
    const float* p_in = pbuf[l & 1];
    float* h_out = hbuf[(l + 1) & 1];
    float* p_out = pbuf[(l + 1) & 1];
    // hs = h @ Wmsg (bf16 WMMA)
    k_rows_gemm<<<2048 / 64, 128, 0, stream>>>(h_in, Wmsgt + l * 128 * 128, hsb, 2048);
    // fused edge + aggregation + node update
    k_edge<<<8 * 256, 128, 0, stream>>>(
        p_in, h_in, hsb,
        W1ac + l * 128 * 192, W2t + l * 128 * 128, W3t + l * 80 * 128,
        W1 + l * 320 * 128, b1 + l * 128, b2 + l * 128,
        Wo1 + l * 128 * 128, bo1 + l * 128,
        Wo2 + l * 128 * 128, bo2 + l * 128,
        Wo3 + l * 128 * 128, bo3 + l * 128,
        p_out, h_out);
  }
  float* vec_out  = (float*)d_out;                 // [B,N,V,3] = 49152
  float* scal_out = vec_out + 8 * 256 * 8 * 3;     // [B,N,F]  = 262144
  k_vec<<<8, 256, 0, stream>>>(pbuf[1], Wfv, vec_out);
  k_rows_gemm<<<2048 / 64, 128, 0, stream>>>(hbuf[1], Wfst, scal_out, 2048);
}
